// TtTrOCRAttention_39367670235201
// MI455X (gfx1250) — matrix operand order repaired and checked by reading, not compiled, not run on hardware
//
#include <hip/hip_runtime.h>

typedef __attribute__((ext_vector_type(16))) _Float16 v16h;
typedef __attribute__((ext_vector_type(8)))  _Float16 v8h;
typedef __attribute__((ext_vector_type(4)))  _Float16 v4h;
typedef __attribute__((ext_vector_type(8)))  float    v8f;

union V16 { v16h v; v8h h[2]; };

#define BATCH 4
#define SEQ   2048
#define EMBED 1024
#define NHEAD 16
#define HDIM  64
#define MROWS (BATCH*SEQ)          // 8192
#define QSCALE 0.125f              // (1/64)^0.5

// ---------------------------------------------------------------------------
// gfx1250 async global->LDS copy (ASYNCcnt-tracked, no VGPR staging).
// vdst = LDS byte address (low 32 bits of generic shared pointer),
// vaddr = 64-bit global address.
// ---------------------------------------------------------------------------
__device__ __forceinline__ void async_copy_b128(void* lds_ptr, const void* gptr) {
    unsigned l = (unsigned)(uintptr_t)lds_ptr;                 // LDS offset
    unsigned long long g = (unsigned long long)(uintptr_t)gptr;
    asm volatile("global_load_async_to_lds_b128 %0, %1, off"
                 :: "v"(l), "v"(g) : "memory");
}

// ---------------------------------------------------------------------------
// fp32 -> fp16 conversion (vectorized x4)
// ---------------------------------------------------------------------------
__global__ void cvt_f32_f16(const float* __restrict__ s, _Float16* __restrict__ d, int n4) {
    int i = blockIdx.x * blockDim.x + threadIdx.x;
    if (i < n4) {
        float4 f = ((const float4*)s)[i];
        v4h h;
        h[0] = (_Float16)f.x; h[1] = (_Float16)f.y;
        h[2] = (_Float16)f.z; h[3] = (_Float16)f.w;
        ((v4h*)d)[i] = h;
    }
}

// ---------------------------------------------------------------------------
// WMMA GEMM:  Y[m][n] = sum_k A[m][k] * W[n][k] + bias[n]
// Double-buffered LDS, async global->LDS copies, 4 WMMAs per wave per k-step.
// Block: 256 thr = 8 waves; block tile 64(M) x 128(N); wave tile 16 x 64.
// mode 0: Q  -> f16 [BH][T][64], * QSCALE
// mode 1: K  -> f16 [BH][T][64]
// mode 2: V  -> f16 [BH][64][T]   (transposed for attention B-fragments)
// mode 3: f32 row-major [8192][1024] (final output)
// ---------------------------------------------------------------------------
__global__ __launch_bounds__(256)
void gemm_wmma(const _Float16* __restrict__ A, const _Float16* __restrict__ W,
               const float* __restrict__ bias, _Float16* __restrict__ out16,
               float* __restrict__ out32, int mode)
{
    __shared__ __align__(16) _Float16 ldsA[2][64 * 40];    // 64 rows x 32 (stride 40)
    __shared__ __align__(16) _Float16 ldsB[2][128 * 40];   // 128 rows x 32

    const int tid  = threadIdx.x;
    const int lane = tid & 31;
    const int wid  = tid >> 5;
    const int wm   = wid & 3;            // M sub-tile (16 rows each)
    const int wn   = wid >> 2;           // N group (64 cols each)
    const int m0   = blockIdx.y * 64;
    const int n0   = blockIdx.x * 128;
    const int hl   = lane >> 4;          // half-wave id
    const int l16  = lane & 15;

    const int arow = tid >> 2, achk = tid & 3;     // A tile: 64 rows x 4 chunks
    const int br0  = tid >> 2,        bc0 = tid & 3;   // B chunk 0
    const int br1  = (tid + 256) >> 2, bc1 = tid & 3;  // B chunk 1

    v8f acc0 = {}, acc1 = {}, acc2 = {}, acc3 = {};

    // per-thread source pointers / LDS slots
    const _Float16* gA = &A[(size_t)(m0 + arow) * EMBED + achk * 8];
    const _Float16* gB0 = &W[(size_t)(n0 + br0) * EMBED + bc0 * 8];
    const _Float16* gB1 = &W[(size_t)(n0 + br1) * EMBED + bc1 * 8];
    const int lA  = arow * 40 + achk * 8;
    const int lB0 = br0 * 40 + bc0 * 8;
    const int lB1 = br1 * 40 + bc1 * 8;

    auto stage = [&](int buf, int k0) __attribute__((always_inline)) {
        async_copy_b128(&ldsA[buf][lA],  gA  + k0);
        async_copy_b128(&ldsB[buf][lB0], gB0 + k0);
        async_copy_b128(&ldsB[buf][lB1], gB1 + k0);
    };
    auto compute = [&](int buf) __attribute__((always_inline)) {
        V16 a;
        a.h[0] = *(const v8h*)&ldsA[buf][(wm * 16 + l16) * 40 + hl * 16];
        a.h[1] = *(const v8h*)&ldsA[buf][(wm * 16 + l16) * 40 + hl * 16 + 8];
        V16 b0, b1, b2, b3;
        const int bn = wn * 64 + l16;
        b0.h[0] = *(const v8h*)&ldsB[buf][(bn +  0) * 40 + hl * 16];
        b0.h[1] = *(const v8h*)&ldsB[buf][(bn +  0) * 40 + hl * 16 + 8];
        b1.h[0] = *(const v8h*)&ldsB[buf][(bn + 16) * 40 + hl * 16];
        b1.h[1] = *(const v8h*)&ldsB[buf][(bn + 16) * 40 + hl * 16 + 8];
        b2.h[0] = *(const v8h*)&ldsB[buf][(bn + 32) * 40 + hl * 16];
        b2.h[1] = *(const v8h*)&ldsB[buf][(bn + 32) * 40 + hl * 16 + 8];
        b3.h[0] = *(const v8h*)&ldsB[buf][(bn + 48) * 40 + hl * 16];
        b3.h[1] = *(const v8h*)&ldsB[buf][(bn + 48) * 40 + hl * 16 + 8];
        acc0 = __builtin_amdgcn_wmma_f32_16x16x32_f16(false, a.v, false, b0.v, (short)0, acc0, false, false);
        acc1 = __builtin_amdgcn_wmma_f32_16x16x32_f16(false, a.v, false, b1.v, (short)0, acc1, false, false);
        acc2 = __builtin_amdgcn_wmma_f32_16x16x32_f16(false, a.v, false, b2.v, (short)0, acc2, false, false);
        acc3 = __builtin_amdgcn_wmma_f32_16x16x32_f16(false, a.v, false, b3.v, (short)0, acc3, false, false);
    };

    stage(0, 0);    // tile 0 -> buffer 0

    const int KSTEPS = EMBED / 32;   // 32, even
#pragma unroll 1
    for (int kk = 0; kk < KSTEPS; kk += 2) {
        // ---- consume buffer 0 (tile kk); DMA tile kk+1 into buffer 1
        stage(1, (kk + 1) * 32);
        asm volatile("s_wait_asynccnt 0x3" ::: "memory");   // tile kk landed
        __syncthreads();
        compute(0);
        __syncthreads();
        // ---- consume buffer 1 (tile kk+1); DMA tile kk+2 into buffer 0
        if (kk + 2 < KSTEPS) {
            stage(0, (kk + 2) * 32);
            asm volatile("s_wait_asynccnt 0x3" ::: "memory");
        } else {
            asm volatile("s_wait_asynccnt 0x0" ::: "memory");
        }
        __syncthreads();
        compute(1);
        __syncthreads();
    }

    // epilogue: C layout row = hl*8 + i, col = l16 within each 16x16 tile
    const int mb = hl * 8;
    const v8f accs[4] = {acc0, acc1, acc2, acc3};
#pragma unroll
    for (int j = 0; j < 4; ++j) {
        int c = n0 + wn * 64 + j * 16 + l16;
        float bv = bias[c];
#pragma unroll
        for (int i = 0; i < 8; ++i) {
            int r = m0 + wm * 16 + mb + i;
            float val = accs[j][i] + bv;
            if (mode == 0) val *= QSCALE;
            if (mode == 3) {
                out32[(size_t)r * EMBED + c] = val;
            } else {
                int b = r >> 11, t = r & (SEQ - 1);
                int h = c >> 6, hd = c & (HDIM - 1);
                int bh = b * NHEAD + h;
                size_t idx = (mode == 2)
                    ? ((size_t)bh * HDIM + hd) * SEQ + t        // V transposed
                    : ((size_t)bh * SEQ + t) * HDIM + hd;       // Q / K
                out16[idx] = (_Float16)val;
            }
        }
    }
}

// ---------------------------------------------------------------------------
// Flash attention: one wave owns 16 query rows of one (batch,head).
// Q,K: [BH][T][64] f16 (Q pre-scaled). Vt: [BH][64][T] f16.
// O:   [8192][1024] f16 (rows b*T+t, cols h*64+hd) -> feeds out-projection.
// Key blocks of 32; online softmax; per-wave LDS staging for P transpose.
// ---------------------------------------------------------------------------
__global__ __launch_bounds__(128)
void attn_wmma(const _Float16* __restrict__ Q, const _Float16* __restrict__ K,
               const _Float16* __restrict__ Vt, _Float16* __restrict__ O)
{
    __shared__ __align__(16) _Float16 ldsP[4][16 * 40];   // per-wave P tile 16x32

    const int lane = threadIdx.x & 31;
    const int wid  = threadIdx.x >> 5;
    const int bh   = blockIdx.x;
    const int q0   = (blockIdx.y * 4 + wid) * 16;
    const int hl   = lane >> 4, l16 = lane & 15;

    const _Float16* Qb = Q  + (size_t)bh * SEQ * HDIM;
    const _Float16* Kb = K  + (size_t)bh * SEQ * HDIM;
    const _Float16* Vb = Vt + (size_t)bh * HDIM * SEQ;

    // Q fragments for K-dim 0..31 and 32..63 (held for the whole loop)
    V16 aQ0, aQ1;
    {
        const _Float16* qr = Qb + (size_t)(q0 + l16) * HDIM;
        aQ0.h[0] = *(const v8h*)(qr + hl * 16);
        aQ0.h[1] = *(const v8h*)(qr + hl * 16 + 8);
        aQ1.h[0] = *(const v8h*)(qr + 32 + hl * 16);
        aQ1.h[1] = *(const v8h*)(qr + 32 + hl * 16 + 8);
    }

    v8f o0 = {}, o1 = {}, o2 = {}, o3 = {};
    float mrow[8], lrow[8];
#pragma unroll
    for (int i = 0; i < 8; ++i) { mrow[i] = -1e30f; lrow[i] = 0.0f; }

    _Float16* P = &ldsP[wid][0];

    for (int kb = 0; kb < SEQ / 32; ++kb) {
        const int tk = kb * 32;

        // ---- S tile (16 q-rows x 32 keys) = Q(16x64) @ K^T(64x32): 4 WMMAs
        const _Float16* kr0 = Kb + (size_t)(tk + l16) * HDIM;        // keys tk..tk+15
        const _Float16* kr1 = Kb + (size_t)(tk + 16 + l16) * HDIM;   // keys tk+16..tk+31
        v16h b00 = *(const v16h*)(kr0 + hl * 16);        // k 0..31
        v16h b01 = *(const v16h*)(kr0 + 32 + hl * 16);   // k 32..63
        v16h b10 = *(const v16h*)(kr1 + hl * 16);
        v16h b11 = *(const v16h*)(kr1 + 32 + hl * 16);
        v8f s0 = {}, s1 = {};
        s0 = __builtin_amdgcn_wmma_f32_16x16x32_f16(false, aQ0.v, false, b00, (short)0, s0, false, false);
        s0 = __builtin_amdgcn_wmma_f32_16x16x32_f16(false, aQ1.v, false, b01, (short)0, s0, false, false);
        s1 = __builtin_amdgcn_wmma_f32_16x16x32_f16(false, aQ0.v, false, b10, (short)0, s1, false, false);
        s1 = __builtin_amdgcn_wmma_f32_16x16x32_f16(false, aQ1.v, false, b11, (short)0, s1, false, false);

        // ---- hoist V B-fragment loads: overlap global latency with softmax ALU
        const _Float16* vbase = Vb + tk + hl * 16;
        v16h bv0 = *(const v16h*)(vbase + (size_t)(0 * 16 + l16) * SEQ);
        v16h bv1 = *(const v16h*)(vbase + (size_t)(1 * 16 + l16) * SEQ);
        v16h bv2 = *(const v16h*)(vbase + (size_t)(2 * 16 + l16) * SEQ);
        v16h bv3 = *(const v16h*)(vbase + (size_t)(3 * 16 + l16) * SEQ);

        // ---- online softmax (rows live across 16-lane halves; xor masks <16 stay in-half)
        float p0[8], p1[8];
#pragma unroll
        for (int i = 0; i < 8; ++i) {
            float mx = fmaxf(s0[i], s1[i]);
            mx = fmaxf(mx, __shfl_xor(mx, 1));
            mx = fmaxf(mx, __shfl_xor(mx, 2));
            mx = fmaxf(mx, __shfl_xor(mx, 4));
            mx = fmaxf(mx, __shfl_xor(mx, 8));
            float mn   = fmaxf(mrow[i], mx);
            float corr = __expf(mrow[i] - mn);
            p0[i] = __expf(s0[i] - mn);
            p1[i] = __expf(s1[i] - mn);
            float rs = p0[i] + p1[i];
            rs += __shfl_xor(rs, 1);
            rs += __shfl_xor(rs, 2);
            rs += __shfl_xor(rs, 4);
            rs += __shfl_xor(rs, 8);
            lrow[i] = lrow[i] * corr + rs;
            mrow[i] = mn;
            o0[i] *= corr; o1[i] *= corr; o2[i] *= corr; o3[i] *= corr;
        }

        // ---- transpose P (C layout -> A layout) via per-wave LDS tile
#pragma unroll
        for (int i = 0; i < 8; ++i) {
            P[(hl * 8 + i) * 40 + l16]      = (_Float16)p0[i];
            P[(hl * 8 + i) * 40 + 16 + l16] = (_Float16)p1[i];
        }
        // wave-local producer/consumer on own LDS tile: DS ops retire in order,
        // just drain the DS counter (no workgroup barrier needed).
        asm volatile("s_wait_dscnt 0x0" ::: "memory");
        V16 aP;
        aP.h[0] = *(const v8h*)&P[l16 * 40 + hl * 16];
        aP.h[1] = *(const v8h*)&P[l16 * 40 + hl * 16 + 8];

        // ---- O += P(16x32) @ V(32x64)
        o0 = __builtin_amdgcn_wmma_f32_16x16x32_f16(false, aP.v, false, bv0, (short)0, o0, false, false);
        o1 = __builtin_amdgcn_wmma_f32_16x16x32_f16(false, aP.v, false, bv1, (short)0, o1, false, false);
        o2 = __builtin_amdgcn_wmma_f32_16x16x32_f16(false, aP.v, false, bv2, (short)0, o2, false, false);
        o3 = __builtin_amdgcn_wmma_f32_16x16x32_f16(false, aP.v, false, bv3, (short)0, o3, false, false);
    }

    // ---- normalize and scatter to [B*T][E] f16 for the output projection
    const int b = bh >> 4, h = bh & (NHEAD - 1);
#pragma unroll
    for (int i = 0; i < 8; ++i) {
        float inv = 1.0f / lrow[i];
        int r = b * SEQ + q0 + hl * 8 + i;
        _Float16* orow = O + (size_t)r * EMBED + h * HDIM;
        orow[0 * 16 + l16] = (_Float16)(o0[i] * inv);
        orow[1 * 16 + l16] = (_Float16)(o1[i] * inv);
        orow[2 * 16 + l16] = (_Float16)(o2[i] * inv);
        orow[3 * 16 + l16] = (_Float16)(o3[i] * inv);
    }
}

// ---------------------------------------------------------------------------
extern "C" void kernel_launch(void* const* d_in, const int* in_sizes, int n_in,
                              void* d_out, int out_size, void* d_ws, size_t ws_size,
                              hipStream_t stream)
{
    const float* hidden = (const float*)d_in[0];
    const float* q_w = (const float*)d_in[1];
    const float* q_b = (const float*)d_in[2];
    const float* k_w = (const float*)d_in[3];
    const float* k_b = (const float*)d_in[4];
    const float* v_w = (const float*)d_in[5];
    const float* v_b = (const float*)d_in[6];
    const float* o_w = (const float*)d_in[7];
    const float* o_b = (const float*)d_in[8];

    const size_t NH = (size_t)MROWS * EMBED;       // 8 Mi elements
    const size_t NW = (size_t)EMBED * EMBED;       // 1 Mi elements
    const size_t NA = (size_t)BATCH * NHEAD * SEQ * HDIM;   // 8 Mi elements

    _Float16* X16  = (_Float16*)d_ws;
    _Float16* Wq16 = X16  + NH;
    _Float16* Wk16 = Wq16 + NW;
    _Float16* Wv16 = Wk16 + NW;
    _Float16* Wo16 = Wv16 + NW;
    _Float16* Q16  = Wo16 + NW;
    _Float16* K16  = Q16  + NA;
    _Float16* Vt16 = K16  + NA;
    _Float16* A16  = Vt16 + NA;    // attention output, [8192][1024] f16

    // fp32 -> fp16 conversions
    cvt_f32_f16<<<(int)(NH / 4 + 255) / 256, 256, 0, stream>>>(hidden, X16, (int)(NH / 4));
    cvt_f32_f16<<<(int)(NW / 4 + 255) / 256, 256, 0, stream>>>(q_w, Wq16, (int)(NW / 4));
    cvt_f32_f16<<<(int)(NW / 4 + 255) / 256, 256, 0, stream>>>(k_w, Wk16, (int)(NW / 4));
    cvt_f32_f16<<<(int)(NW / 4 + 255) / 256, 256, 0, stream>>>(v_w, Wv16, (int)(NW / 4));
    cvt_f32_f16<<<(int)(NW / 4 + 255) / 256, 256, 0, stream>>>(o_w, Wo16, (int)(NW / 4));

    // QKV projections (WMMA)
    dim3 gg(EMBED / 128, MROWS / 64);
    gemm_wmma<<<gg, 256, 0, stream>>>(X16, Wq16, q_b, Q16,  nullptr, 0);
    gemm_wmma<<<gg, 256, 0, stream>>>(X16, Wk16, k_b, K16,  nullptr, 1);
    gemm_wmma<<<gg, 256, 0, stream>>>(X16, Wv16, v_b, Vt16, nullptr, 2);

    // Flash attention (WMMA): 64 (b,h) x 32 blocks x 4 waves x 16 q-rows
    attn_wmma<<<dim3(BATCH * NHEAD, SEQ / 16 / 4), 128, 0, stream>>>(Q16, K16, Vt16, A16);

    // output projection -> fp32 d_out (WMMA)
    gemm_wmma<<<gg, 256, 0, stream>>>(A16, Wo16, o_b, nullptr, (float*)d_out, 3);
}
